// GausSplatingHead_52742198395574
// MI455X (gfx1250) — matrix operand order repaired and checked by reading, not compiled, not run on hardware
//
#include <hip/hip_runtime.h>
#include <hip/hip_bf16.h>

typedef __attribute__((ext_vector_type(16))) _Float16 v16h;
typedef __attribute__((ext_vector_type(8)))  float    v8f;

#define RHh 40
#define RWw 64
#define PIX 2560      // RH*RW
#define FD  17
#define SAMQ 4096     // 64*64
#define NMASK 16

// ------------------------------------------------------------------ utils

__global__ void zero_kernel(float* out){
  if (threadIdx.x == 0) out[0] = 0.f;
}

// cast weights f32 -> f16 once per launch (nSrc valid rows, pad to nDst rows of K with zeros)
__global__ __launch_bounds__(256) void castw_kernel(
    const float* __restrict__ src, _Float16* __restrict__ dst, int nValid, int total)
{
  int i = blockIdx.x * 256 + threadIdx.x;
  if (i < total) dst[i] = (i < nValid) ? (_Float16)src[i] : (_Float16)0.f;
}

// ------------------------------------------------------------------ gaussian prep (per camera)

__global__ __launch_bounds__(256) void prep_kernel(
    const float* __restrict__ xyz, const float* __restrict__ lscale,
    const float* __restrict__ rots, const float* __restrict__ opac,
    const float* __restrict__ vm, const float* __restrict__ intr,
    float* __restrict__ params, unsigned long long* __restrict__ keys, int n)
{
  int i = blockIdx.x * 256 + threadIdx.x;
  if (i >= n) return;
  float X = xyz[i*3+0], Y = xyz[i*3+1], Z = xyz[i*3+2];
  float Rw[9] = { vm[0],vm[1],vm[2], vm[4],vm[5],vm[6], vm[8],vm[9],vm[10] };
  float t0 = vm[3], t1 = vm[7], t2 = vm[11];
  float cxp = Rw[0]*X + Rw[1]*Y + Rw[2]*Z + t0;
  float cyp = Rw[3]*X + Rw[4]*Y + Rw[5]*Z + t1;
  float cz  = Rw[6]*X + Rw[7]*Y + Rw[8]*Z + t2;
  float fx = intr[0], fy = intr[1], ppx = intr[2], ppy = intr[3];
  float zc = fmaxf(cz, 0.001f);
  float u = fx * cxp / zc + ppx;
  float v = fy * cyp / zc + ppy;

  float qw = rots[i*4+0], qx = rots[i*4+1], qy = rots[i*4+2], qz = rots[i*4+3];
  float qn = rsqrtf(qw*qw + qx*qx + qy*qy + qz*qz);
  qw *= qn; qx *= qn; qy *= qn; qz *= qn;
  float Rq[9] = {
    1.f-2.f*(qy*qy+qz*qz), 2.f*(qx*qy-qw*qz),     2.f*(qx*qz+qw*qy),
    2.f*(qx*qy+qw*qz),     1.f-2.f*(qx*qx+qz*qz), 2.f*(qy*qz-qw*qx),
    2.f*(qx*qz-qw*qy),     2.f*(qy*qz+qw*qx),     1.f-2.f*(qx*qx+qy*qy)
  };
  float s0 = __expf(lscale[i*3+0]);
  float s1 = __expf(lscale[i*3+1]);
  float s2 = __expf(lscale[i*3+2]);
  float M[9] = { Rq[0]*s0, Rq[1]*s1, Rq[2]*s2,
                 Rq[3]*s0, Rq[4]*s1, Rq[5]*s2,
                 Rq[6]*s0, Rq[7]*s1, Rq[8]*s2 };
  float C3[9], Tm[9], Cw[9];
#pragma unroll
  for (int r = 0; r < 3; ++r)
#pragma unroll
    for (int c = 0; c < 3; ++c)
      C3[r*3+c] = M[r*3+0]*M[c*3+0] + M[r*3+1]*M[c*3+1] + M[r*3+2]*M[c*3+2];
#pragma unroll
  for (int r = 0; r < 3; ++r)
#pragma unroll
    for (int c = 0; c < 3; ++c)
      Tm[r*3+c] = Rw[r*3+0]*C3[0*3+c] + Rw[r*3+1]*C3[1*3+c] + Rw[r*3+2]*C3[2*3+c];
#pragma unroll
  for (int r = 0; r < 3; ++r)
#pragma unroll
    for (int c = 0; c < 3; ++c)
      Cw[r*3+c] = Tm[r*3+0]*Rw[c*3+0] + Tm[r*3+1]*Rw[c*3+1] + Tm[r*3+2]*Rw[c*3+2];

  float iz  = 1.f / zc;
  float J00 = fx*iz, J02 = -fx*cxp*iz*iz;
  float J11 = fy*iz, J12 = -fy*cyp*iz*iz;
  float K00 = J00*Cw[0] + J02*Cw[6];
  float K01 = J00*Cw[1] + J02*Cw[7];
  float K02 = J00*Cw[2] + J02*Cw[8];
  float K11 = J11*Cw[4] + J12*Cw[7];
  float K12 = J11*Cw[5] + J12*Cw[8];
  float c00 = K00*J00 + K02*J02 + 0.3f;
  float c01 = K01*J11 + K02*J12;
  float c11 = K11*J11 + K12*J12 + 0.3f;
  float det = c00*c11 - c01*c01;
  float Ai = c11/det, Bi = -c01/det, Ci = c00/det;
  float opv = (cz > 0.2f) ? opac[i] : 0.f;

  params[i*6+0] = u;  params[i*6+1] = v;
  params[i*6+2] = Ai; params[i*6+3] = Bi;
  params[i*6+4] = Ci; params[i*6+5] = opv;

  unsigned int ub = __float_as_uint(cz);
  ub = (ub & 0x80000000u) ? ~ub : (ub | 0x80000000u);
  keys[i] = ((unsigned long long)ub << 32) | (unsigned int)i;   // stable: ties break by index
}

// ------------------------------------------------------------------ stable O(N^2) rank sort (N = 12800, 50 blocks)

__global__ __launch_bounds__(256) void rank_kernel(
    const unsigned long long* __restrict__ keys, int* __restrict__ sidx, int n)
{
  __shared__ unsigned long long kk[256];
  int i = blockIdx.x * 256 + threadIdx.x;
  unsigned long long mine = keys[i];
  int rank = 0;
  for (int base = 0; base < n; base += 256){
    kk[threadIdx.x] = keys[base + threadIdx.x];
    __syncthreads();
#pragma unroll 8
    for (int j = 0; j < 256; ++j) rank += (kk[j] < mine) ? 1 : 0;
    __syncthreads();
  }
  sidx[rank] = i;
}

// ------------------------------------------------------------------ alpha compositing (1 thread = 1 pixel, LDS gaussian tiles)

__global__ __launch_bounds__(256) void composite_kernel(
    const float* __restrict__ params, const int* __restrict__ sidx,
    const float* __restrict__ feats, float* __restrict__ fmap, int n)
{
  __shared__ float sp[128*6];
  __shared__ float sf[128*FD];
  int p  = blockIdx.x * 256 + threadIdx.x;
  float px = (float)(p & 63);
  float py = (float)(p >> 6);
  float T = 1.f;
  float acc[FD];
#pragma unroll
  for (int ch = 0; ch < FD; ++ch) acc[ch] = 0.f;

  for (int base = 0; base < n; base += 128){
    __builtin_prefetch(sidx + base + 128, 0, 1);
    for (int e = threadIdx.x; e < 128*(6+FD); e += 256){
      int g = e / (6+FD);
      int f = e - g*(6+FD);
      int gi = sidx[base + g];
      if (f < 6) sp[g*6 + f] = params[gi*6 + f];
      else       sf[g*FD + (f-6)] = feats[gi*FD + (f-6)];
    }
    __syncthreads();
    for (int g = 0; g < 128; ++g){
      float u  = sp[g*6+0], v  = sp[g*6+1];
      float Ai = sp[g*6+2], Bi = sp[g*6+3], Ci = sp[g*6+4], op = sp[g*6+5];
      float dx = u - px, dy = v - py;
      float pw = -0.5f*(Ai*dx*dx + Ci*dy*dy) - Bi*dx*dy;
      float al = fminf(0.99f, op * __expf(fminf(pw, 0.f)));
      float w  = al * T;
#pragma unroll
      for (int ch = 0; ch < FD; ++ch) acc[ch] += w * sf[g*FD + ch];
      T *= (1.f - al);
    }
    __syncthreads();
  }
#pragma unroll
  for (int ch = 0; ch < FD; ++ch) fmap[p*FD + ch] = acc[ch];
}

// ------------------------------------------------------------------ WMMA fragment loaders (CDNA5 16x16x32 f16 layouts)

__device__ __forceinline__ v16h load_a_frag(const _Float16* buf, int stride, int row, int hi, int kofs){
  alignas(32) _Float16 e[16];
#pragma unroll
  for (int j = 0; j < 8; ++j){
    int kb = kofs + 2*j + 16*hi;
    e[2*j]   = buf[row*stride + kb];
    e[2*j+1] = buf[row*stride + kb + 1];
  }
  v16h r; __builtin_memcpy(&r, e, sizeof(r));
  return r;
}

// B fragment from pre-cast f16 weights: 16 contiguous halves -> 2x b128 loads, no cvt
__device__ __forceinline__ v16h load_b_frag_h(const _Float16* __restrict__ W, int K, int n, int kofs, int hi){
  alignas(32) _Float16 e[16];
  const _Float16* p = W + n*K + kofs + 16*hi;
#pragma unroll
  for (int i = 0; i < 16; ++i) e[i] = p[i];
  v16h r; __builtin_memcpy(&r, e, sizeof(r));
  return r;
}

// ------------------------------------------------------------------ fused 3-layer MLP (LN + leaky) via v_wmma_f32_16x16x32_f16
// mode 0: nearest-resized sam_embd gather (rows = 2560); mode 1: sam_features identity (rows = 4096)

__global__ __launch_bounds__(256) void mlp_kernel(
    const float* __restrict__ src, int mode,
    const _Float16* __restrict__ W1h, const float* __restrict__ b1,
    const float* __restrict__ g1, const float* __restrict__ be1,
    const _Float16* __restrict__ W2h, const float* __restrict__ b2,
    const float* __restrict__ g2, const float* __restrict__ be2,
    const _Float16* __restrict__ W3h, const float* __restrict__ b3,
    float* __restrict__ out)
{
  __shared__ _Float16 Xs[128*32];   // staged K-chunk of activations
  __shared__ float    Hs[128*64];   // f32 hidden for LN
  __shared__ _Float16 Hh[128*64];   // f16 hidden for next GEMM

  const int tid   = threadIdx.x;
  const int lane  = tid & 31;
  const int wv    = tid >> 5;
  const int lr    = lane & 15;
  const int hi    = lane >> 4;
  const int mbase = wv * 16;
  const int rowBase = blockIdx.x * 128;

  // staging role: thread -> (row, k-half)
  const int sr = tid & 127;
  const int sh = tid >> 7;
  int gofs;
  {
    int row = rowBase + sr;
    if (mode == 0){ int h = row >> 6, w = row & 63; gofs = ((h*64)/40)*64 + w; }
    else gofs = row;
  }

  // ---------------- layer 1: K=256 ----------------
  v8f c1[4] = {};
  for (int kc = 0; kc < 8; ++kc){
#pragma unroll
    for (int i = 0; i < 16; ++i){
      int k = kc*32 + sh*16 + i;
      Xs[sr*32 + sh*16 + i] = (_Float16)src[k*4096 + gofs];
    }
    __syncthreads();
    v16h a = load_a_frag(Xs, 32, mbase + lr, hi, 0);
#pragma unroll
    for (int nt = 0; nt < 4; ++nt){
      v16h bf = load_b_frag_h(W1h, 256, nt*16 + lr, kc*32, hi);
      c1[nt] = __builtin_amdgcn_wmma_f32_16x16x32_f16(false, a, false, bf, (short)0, c1[nt], false, false);
    }
    __syncthreads();
  }
#pragma unroll
  for (int nt = 0; nt < 4; ++nt){
    int n = nt*16 + lr;
    float bv = b1[n];
    const float* cp = (const float*)&c1[nt];
#pragma unroll
    for (int j = 0; j < 8; ++j) Hs[(mbase + j + 8*hi)*64 + n] = cp[j] + bv;
  }
  __syncthreads();
  if (tid < 128){
    float mu = 0.f;
#pragma unroll 8
    for (int k = 0; k < 64; ++k) mu += Hs[tid*64 + k];
    mu *= (1.f/64.f);
    float var = 0.f;
#pragma unroll 8
    for (int k = 0; k < 64; ++k){ float d = Hs[tid*64 + k] - mu; var += d*d; }
    var *= (1.f/64.f);
    float inv = rsqrtf(var + 1e-5f);
#pragma unroll 8
    for (int k = 0; k < 64; ++k){
      float vl = (Hs[tid*64 + k] - mu)*inv*g1[k] + be1[k];
      vl = (vl >= 0.f) ? vl : 0.01f*vl;
      Hh[tid*64 + k] = (_Float16)vl;
    }
  }
  __syncthreads();

  // ---------------- layer 2: K=64 ----------------
  v8f c2[4] = {};
#pragma unroll
  for (int kc = 0; kc < 2; ++kc){
    v16h a = load_a_frag(Hh, 64, mbase + lr, hi, kc*32);
#pragma unroll
    for (int nt = 0; nt < 4; ++nt){
      v16h bf = load_b_frag_h(W2h, 64, nt*16 + lr, kc*32, hi);
      c2[nt] = __builtin_amdgcn_wmma_f32_16x16x32_f16(false, a, false, bf, (short)0, c2[nt], false, false);
    }
  }
  __syncthreads();
#pragma unroll
  for (int nt = 0; nt < 4; ++nt){
    int n = nt*16 + lr;
    float bv = b2[n];
    const float* cp = (const float*)&c2[nt];
#pragma unroll
    for (int j = 0; j < 8; ++j) Hs[(mbase + j + 8*hi)*64 + n] = cp[j] + bv;
  }
  __syncthreads();
  if (tid < 128){
    float mu = 0.f;
#pragma unroll 8
    for (int k = 0; k < 64; ++k) mu += Hs[tid*64 + k];
    mu *= (1.f/64.f);
    float var = 0.f;
#pragma unroll 8
    for (int k = 0; k < 64; ++k){ float d = Hs[tid*64 + k] - mu; var += d*d; }
    var *= (1.f/64.f);
    float inv = rsqrtf(var + 1e-5f);
#pragma unroll 8
    for (int k = 0; k < 64; ++k){
      float vl = (Hs[tid*64 + k] - mu)*inv*g2[k] + be2[k];
      vl = (vl >= 0.f) ? vl : 0.01f*vl;
      Hh[tid*64 + k] = (_Float16)vl;
    }
  }
  __syncthreads();

  // ---------------- layer 3: K=64, N=17 (W3h zero-padded to 32 rows) ----------------
  v8f c3[2] = {};
#pragma unroll
  for (int kc = 0; kc < 2; ++kc){
    v16h a = load_a_frag(Hh, 64, mbase + lr, hi, kc*32);
#pragma unroll
    for (int nt = 0; nt < 2; ++nt){
      v16h bf = load_b_frag_h(W3h, 64, nt*16 + lr, kc*32, hi);
      c3[nt] = __builtin_amdgcn_wmma_f32_16x16x32_f16(false, a, false, bf, (short)0, c3[nt], false, false);
    }
  }
#pragma unroll
  for (int nt = 0; nt < 2; ++nt){
    int n = nt*16 + lr;
    if (n < 17){
      float bv = b3[n];
      const float* cp = (const float*)&c3[nt];
#pragma unroll
      for (int j = 0; j < 8; ++j){
        int grow = rowBase + mbase + j + 8*hi;
        out[grow*FD + n] = cp[j] + bv;
      }
    }
  }
}

// ------------------------------------------------------------------ |fmap - sed| mean

__global__ __launch_bounds__(256) void l1_kernel(
    const float* __restrict__ a, const float* __restrict__ b,
    float* __restrict__ out, int n, float scale)
{
  __shared__ float red[256];
  float s = 0.f;
  for (int i = blockIdx.x*256 + threadIdx.x; i < n; i += gridDim.x*256)
    s += fabsf(a[i] - b[i]);
  red[threadIdx.x] = s;
  __syncthreads();
  for (int st = 128; st > 0; st >>= 1){
    if (threadIdx.x < st) red[threadIdx.x] += red[threadIdx.x + st];
    __syncthreads();
  }
  if (threadIdx.x == 0) atomicAdd(out, red[0]*scale);
}

// ------------------------------------------------------------------ mask prototypes (one block per mask)

__global__ __launch_bounds__(256) void proto_kernel(
    const float* __restrict__ masks, const float* __restrict__ ldf, float* __restrict__ proto)
{
  __shared__ float red[18][256];
  __shared__ float res[18];
  int m = blockIdx.x;
  float acc[18];
#pragma unroll
  for (int q = 0; q < 18; ++q) acc[q] = 0.f;
  for (int p = threadIdx.x; p < SAMQ; p += 256){
    int h = p >> 6, w = p & 63;
    float mv = masks[m*40000 + ((h*200)/64)*200 + ((w*200)/64)];
    if (mv != 0.f){
      acc[17] += mv;
#pragma unroll
      for (int ch = 0; ch < FD; ++ch) acc[ch] += mv * ldf[p*FD + ch];
    }
  }
#pragma unroll
  for (int q = 0; q < 18; ++q) red[q][threadIdx.x] = acc[q];
  __syncthreads();
  if (threadIdx.x < 18){
    float s = 0.f;
    for (int t = 0; t < 256; ++t) s += red[threadIdx.x][t];
    res[threadIdx.x] = s;
  }
  __syncthreads();
  if (threadIdx.x < 17)
    proto[m*FD + threadIdx.x] = res[threadIdx.x] / fmaxf(res[17], 1e-6f);
}

// ------------------------------------------------------------------ BCE with bilinear mask resize

__global__ __launch_bounds__(256) void bce_kernel(
    const float* __restrict__ masks, const float* __restrict__ proto,
    const float* __restrict__ fmap, float* __restrict__ out, float scale)
{
  __shared__ float red[256];
  int idx = blockIdx.x*256 + threadIdx.x;
  int m = idx / PIX;
  int p = idx - m*PIX;
  int h = p >> 6, w = p & 63;
  float logit = 0.f;
#pragma unroll
  for (int ch = 0; ch < FD; ++ch) logit += proto[m*FD + ch] * fmap[p*FD + ch];
  float prob = 1.f / (1.f + __expf(-logit));

  float srr = (h + 0.5f)*5.0f - 0.5f;               // 200/40
  int r0 = (int)floorf(srr); r0 = r0 < 0 ? 0 : (r0 > 199 ? 199 : r0);
  int r1 = (r0 + 1 > 199) ? 199 : r0 + 1;
  float wr = srr - (float)r0; wr = wr < 0.f ? 0.f : (wr > 1.f ? 1.f : wr);
  float scc = (w + 0.5f)*3.125f - 0.5f;             // 200/64
  int cc0 = (int)floorf(scc); cc0 = cc0 < 0 ? 0 : (cc0 > 199 ? 199 : cc0);
  int cc1 = (cc0 + 1 > 199) ? 199 : cc0 + 1;
  float wc = scc - (float)cc0; wc = wc < 0.f ? 0.f : (wc > 1.f ? 1.f : wc);
  const float* Mm = masks + m*40000;
  float left  = Mm[r0*200 + cc0]*(1.f-wr) + Mm[r1*200 + cc0]*wr;
  float right = Mm[r0*200 + cc1]*(1.f-wr) + Mm[r1*200 + cc1]*wr;
  float frm = left*(1.f-wc) + right*wc;
  frm = (frm <= 0.5f) ? 0.f : frm;
  float bce = frm*__logf(prob + 1e-8f) + (1.f-frm)*__logf(1.f-prob + 1e-8f);

  red[threadIdx.x] = -bce;
  __syncthreads();
  for (int st = 128; st > 0; st >>= 1){
    if (threadIdx.x < st) red[threadIdx.x] += red[threadIdx.x + st];
    __syncthreads();
  }
  if (threadIdx.x == 0) atomicAdd(out, red[0]*scale);
}

// ------------------------------------------------------------------ launch

extern "C" void kernel_launch(void* const* d_in, const int* in_sizes, int n_in,
                              void* d_out, int out_size, void* d_ws, size_t ws_size,
                              hipStream_t stream)
{
  const float* vf    = (const float*)d_in[0];
  const float* opac  = (const float*)d_in[1];
  const float* embd  = (const float*)d_in[2];
  const float* sfeat = (const float*)d_in[3];
  const float* masks = (const float*)d_in[4];
  const float* vm    = (const float*)d_in[5];
  const float* intr  = (const float*)d_in[6];
  const float* xyz   = (const float*)d_in[7];
  const float* scal  = (const float*)d_in[8];
  const float* rots  = (const float*)d_in[9];
  const float* W1 = (const float*)d_in[10];
  const float* b1 = (const float*)d_in[11];
  const float* g1 = (const float*)d_in[12];
  const float* be1= (const float*)d_in[13];
  const float* W2 = (const float*)d_in[14];
  const float* b2 = (const float*)d_in[15];
  const float* g2 = (const float*)d_in[16];
  const float* be2= (const float*)d_in[17];
  const float* W3 = (const float*)d_in[18];
  const float* b3 = (const float*)d_in[19];
  float* out = (float*)d_out;
  float* ws  = (float*)d_ws;

  const int N = in_sizes[7] / 3;          // 12800

  // global (camera-independent) region: f16 weights
  size_t goff = 0;
  const size_t o_w1h = goff; goff += (64*256)/2;        // 16384 halves = 8192 float slots
  const size_t o_w2h = goff; goff += (64*64)/2;         // 4096 halves  = 2048 float slots
  const size_t o_w3h = goff; goff += (32*64)/2;         // padded to 32 rows = 1024 float slots... (2048 halves)
  goff = (goff + 63) & ~(size_t)63;

  _Float16* W1h = (_Float16*)(ws + o_w1h);
  _Float16* W2h = (_Float16*)(ws + o_w2h);
  _Float16* W3h = (_Float16*)(ws + o_w3h);

  // per-camera regions
  size_t per = 0;
  const size_t o_params = per; per += (size_t)6*N;
  const size_t o_keys   = per; per += (size_t)2*N;      // ull
  const size_t o_sidx   = per; per += (size_t)N;        // int
  const size_t o_fmap   = per; per += (size_t)PIX*FD;
  const size_t o_sed    = per; per += (size_t)PIX*FD;
  const size_t o_ldf    = per; per += (size_t)SAMQ*FD;
  const size_t o_proto  = per; per += 512;
  per = (per + 63) & ~(size_t)63;

  zero_kernel<<<1, 32, 0, stream>>>(out);
  castw_kernel<<<(64*256 + 255)/256, 256, 0, stream>>>(W1, W1h, 64*256, 64*256);
  castw_kernel<<<(64*64  + 255)/256, 256, 0, stream>>>(W2, W2h, 64*64,  64*64);
  castw_kernel<<<(32*64  + 255)/256, 256, 0, stream>>>(W3, W3h, 17*64,  32*64); // zero-pad rows 17..31

  for (int c = 0; c < 2; ++c){
    float* base = ws + goff + (size_t)c * per;
    float* params = base + o_params;
    unsigned long long* keys = (unsigned long long*)(base + o_keys);
    int* sidx   = (int*)(base + o_sidx);
    float* fmap = base + o_fmap;
    float* sed  = base + o_sed;
    float* ldf  = base + o_ldf;
    float* prt  = base + o_proto;

    prep_kernel<<<(N + 255)/256, 256, 0, stream>>>(xyz, scal, rots, opac,
                                                   vm + c*16, intr + c*4, params, keys, N);
    rank_kernel<<<N/256, 256, 0, stream>>>(keys, sidx, N);
    composite_kernel<<<PIX/256, 256, 0, stream>>>(params, sidx, vf, fmap, N);

    mlp_kernel<<<PIX/128, 256, 0, stream>>>(embd + (size_t)c*256*4096, 0,
                                            W1h,b1,g1,be1, W2h,b2,g2,be2, W3h,b3, sed);
    l1_kernel<<<40, 256, 0, stream>>>(fmap, sed, out, PIX*FD, 1.f/((float)(PIX*FD)) * 0.5f);

    mlp_kernel<<<SAMQ/128, 256, 0, stream>>>(sfeat + (size_t)c*256*4096, 1,
                                             W1h,b1,g1,be1, W2h,b2,g2,be2, W3h,b3, ldf);
    proto_kernel<<<NMASK, 256, 0, stream>>>(masks + (size_t)c*NMASK*40000, ldf, prt);
    bce_kernel<<<(NMASK*PIX)/256, 256, 0, stream>>>(masks + (size_t)c*NMASK*40000, prt, fmap, out,
                                                    1.f/((float)(NMASK*PIX)) * 0.5f);
  }
}
